// SSMBranch_54400055771540
// MI455X (gfx1250) — matrix operand order repaired and checked by reading, not compiled, not run on hardware
//
#include <hip/hip_runtime.h>
#include <cmath>

typedef __bf16 bf16;
typedef __attribute__((ext_vector_type(16))) __bf16 v16bf;
typedef __attribute__((ext_vector_type(8)))  float  v8f;

#define NB  4
#define NL  4096
#define ND  1024
#define NSTATE 16
#define DTRANK 64
#define NROWS (NB * NL)           // 16384
#define NXP (DTRANK + 2 * NSTATE) // 96
#define BROW 40                   // LDS row pitch in bf16 (80B: bank-conflict-free, 16B aligned)

// ---------------------------------------------------------------------------
// WMMA helper: D = A(16x32 bf16) * B(32x16 bf16) + C(16x16 f32)
// ---------------------------------------------------------------------------
static __device__ __forceinline__ v8f wmma_bf16(v16bf a, v16bf b, v8f c) {
  return __builtin_amdgcn_wmma_f32_16x16x32_bf16(
      /*neg_a=*/false, a, /*neg_b=*/false, b,
      /*c_mod=*/(short)0, c, /*reuse_a=*/false, /*reuse_b=*/false);
}

// A fragment (16x32, row-major source, lda elements). Layout matching B:
// lane&15 -> M, elements e -> K = k0 + 16*(lane>=16) + e;
// per lane this is one contiguous 16-element (32B) run.
static __device__ __forceinline__ v16bf load_A_frag(const bf16* A, int lda,
                                                    int m0, int k0, int lane) {
  const int m  = m0 + (lane & 15);
  const int kh = (lane >> 4) << 4;
  const bf16* p = A + (size_t)m * lda + k0 + kh;
  v16bf a;
#pragma unroll
  for (int e = 0; e < 16; ++e) a[e] = p[e];
  return a;
}

// B fragment (32x16) from K-major (pre-transposed) weights WT[N][K]:
// lane&15 -> N, elements e -> K = k0 + 16*(lane>=16) + e  (contiguous 32B run).
static __device__ __forceinline__ v16bf load_B_frag(const bf16* WT, int ldk,
                                                    int k0, int n0, int lane) {
  const int n  = n0 + (lane & 15);
  const int kh = (lane >> 4) << 4;
  const bf16* p = WT + (size_t)n * ldk + k0 + kh;
  v16bf b;
#pragma unroll
  for (int e = 0; e < 16; ++e) b[e] = p[e];
  return b;
}

// B fragment from an LDS-staged tile: buf[n_local][k_elem], row pitch BROW.
static __device__ __forceinline__ v16bf load_B_lds(const bf16* buf, int n0loc,
                                                   int lane) {
  const bf16* p = buf + (size_t)(n0loc + (lane & 15)) * BROW + ((lane >> 4) << 4);
  v16bf b;
#pragma unroll
  for (int e = 0; e < 16; ++e) b[e] = p[e];
  return b;
}

// C/D 16x16 f32 layout: VGPR r -> M = r + 8*(lane>=16), N = lane&15.
static __device__ __forceinline__ void store_tile(float* C, int ldc, int m0,
                                                  int n0, int lane, v8f acc,
                                                  float bias) {
  const int m = m0 + ((lane >> 4) << 3);
  const int n = n0 + (lane & 15);
#pragma unroll
  for (int r = 0; r < 8; ++r) C[(size_t)(m + r) * ldc + n] = acc[r] + bias;
}

// Async stage of one 32(K)x64(N) bf16 B tile into LDS (ASYNCcnt-tracked).
// 256 chunks of 16B, 128 threads -> 2 GLOBAL_LOAD_ASYNC_TO_LDS_B128 each.
static __device__ __forceinline__ void stage_B_async(const bf16* WT, int n0,
                                                     int k0, bf16* buf,
                                                     int tid) {
#pragma unroll
  for (int t = 0; t < 2; ++t) {
    const int chunk = tid + t * 128;
    const int nl = chunk >> 2;           // 0..63
    const int c  = chunk & 3;            // 16B chunk within 64B row
    const unsigned lds  = (unsigned)(uintptr_t)(buf + (size_t)nl * BROW + c * 8);
    const unsigned goff = (unsigned)(((n0 + nl) * ND + k0 + c * 8) * 2);
    asm volatile("global_load_async_to_lds_b128 %0, %1, %2"
                 :: "v"(lds), "v"(goff), "s"(WT) : "memory");
  }
}

static __device__ __forceinline__ void wait_async0() {
  asm volatile("s_wait_asynccnt 0x0" ::: "memory");
}

// ---------------------------------------------------------------------------
// Weight conversion: f32 row-major [K][N] -> bf16 K-major (transposed) [N][K]
// ---------------------------------------------------------------------------
__global__ void transpose_to_bf16(const float* __restrict__ src,
                                  bf16* __restrict__ dst, int K, int N) {
  int idx = blockIdx.x * blockDim.x + threadIdx.x;
  if (idx < K * N) {
    int k = idx / N, n = idx - k * N;
    dst[(size_t)n * K + k] = (bf16)src[idx];
  }
}

// ---------------------------------------------------------------------------
// LayerNorm over D=1024, output bf16 (feeds GEMM1 A operand)
// ---------------------------------------------------------------------------
__global__ __launch_bounds__(256) void layernorm_bf16_kernel(
    const float* __restrict__ f, const float* __restrict__ g,
    const float* __restrict__ be, bf16* __restrict__ out) {
  __shared__ float red[256];
  const int row = blockIdx.x;
  const float* xr = f + (size_t)row * ND;
  float v[4], s = 0.f, ss = 0.f;
#pragma unroll
  for (int j = 0; j < 4; ++j) {
    v[j] = xr[threadIdx.x + j * 256];
    s += v[j]; ss += v[j] * v[j];
  }
  red[threadIdx.x] = s; __syncthreads();
  for (int o = 128; o > 0; o >>= 1) {
    if (threadIdx.x < o) red[threadIdx.x] += red[threadIdx.x + o];
    __syncthreads();
  }
  const float mean = red[0] * (1.f / ND); __syncthreads();
  red[threadIdx.x] = ss; __syncthreads();
  for (int o = 128; o > 0; o >>= 1) {
    if (threadIdx.x < o) red[threadIdx.x] += red[threadIdx.x + o];
    __syncthreads();
  }
  const float var  = red[0] * (1.f / ND) - mean * mean;
  const float rstd = rsqrtf(var + 1e-5f);
#pragma unroll
  for (int j = 0; j < 4; ++j) {
    const int col = threadIdx.x + j * 256;
    out[(size_t)row * ND + col] = (bf16)((v[j] - mean) * rstd * g[col] + be[col]);
  }
}

// ---------------------------------------------------------------------------
// GEMM1: x_pre[16384x1024] = fn_bf16 @ Wx_bf16 + bx
// 128 threads = 2x2 waves; each wave owns a 32x32 register tile (4 wmma accs).
// B tile (32x64) is double-buffered in LDS via GLOBAL_LOAD_ASYNC_TO_LDS_B128;
// A fragments stream directly from global (b128, L2-resident rows).
// ---------------------------------------------------------------------------
__global__ __launch_bounds__(128) void gemm1_kernel(
    const bf16* __restrict__ A, const bf16* __restrict__ WT,
    const float* __restrict__ bias, float* __restrict__ C) {
  __shared__ __attribute__((aligned(16))) bf16 sB[2][64 * BROW];
  const int tid  = threadIdx.x;
  const int lane = tid & 31;
  const int wave = tid >> 5;
  const int m0 = blockIdx.y * 64 + (wave >> 1) * 32;
  const int nblk = blockIdx.x * 64;
  const int nloc = (wave & 1) * 32;         // wave's N offset inside LDS tile
  const int n0 = nblk + nloc;

  v8f acc[2][2] = {};
  stage_B_async(WT, nblk, 0, sB[0], tid);
  wait_async0();
  __syncthreads();

  int cur = 0;
  for (int k0 = 0; k0 < ND; k0 += 32) {
    if (k0 + 32 < ND) stage_B_async(WT, nblk, k0 + 32, sB[cur ^ 1], tid);
    v16bf a0 = load_A_frag(A, ND, m0,      k0, lane);
    v16bf a1 = load_A_frag(A, ND, m0 + 16, k0, lane);
    v16bf b0 = load_B_lds(sB[cur], nloc,      lane);
    v16bf b1 = load_B_lds(sB[cur], nloc + 16, lane);
    acc[0][0] = wmma_bf16(a0, b0, acc[0][0]);
    acc[0][1] = wmma_bf16(a0, b1, acc[0][1]);
    acc[1][0] = wmma_bf16(a1, b0, acc[1][0]);
    acc[1][1] = wmma_bf16(a1, b1, acc[1][1]);
    wait_async0();
    __syncthreads();
    cur ^= 1;
  }
#pragma unroll
  for (int i = 0; i < 2; ++i)
#pragma unroll
    for (int j = 0; j < 2; ++j) {
      const int nn = n0 + j * 16 + (lane & 15);
      store_tile(C, ND, m0 + i * 16, n0 + j * 16, lane, acc[i][j], bias[nn]);
    }
}

// ---------------------------------------------------------------------------
// Depthwise causal conv (K=4) + bias + SiLU; emits f32 (scan) and bf16 (GEMM2)
// ---------------------------------------------------------------------------
__global__ __launch_bounds__(256) void conv_silu_kernel(
    const float* __restrict__ xpre, const float* __restrict__ cw,
    const float* __restrict__ cb, float* __restrict__ xf,
    bf16* __restrict__ xb) {
  const size_t total = (size_t)NROWS * ND;
  for (size_t idx = (size_t)blockIdx.x * 256 + threadIdx.x; idx < total;
       idx += (size_t)gridDim.x * 256) {
    const int d = (int)(idx & (ND - 1));
    const int l = (int)((idx >> 10) & (NL - 1));
    const size_t brow = idx - (size_t)l * ND - (size_t)d; // (b*NL)*ND
    float acc = cb[d];
#pragma unroll
    for (int k = 0; k < 4; ++k) {
      const int ls = l - 3 + k;
      if (ls >= 0)
        acc = fmaf(cw[(d << 2) + k], xpre[brow + (size_t)ls * ND + d], acc);
    }
    const float v = acc / (1.f + __expf(-acc)); // SiLU
    xf[idx] = v;
    xb[idx] = (bf16)v;
  }
}

// ---------------------------------------------------------------------------
// GEMM2: x_dbl[16384x96] = x_bf16 @ Wxp_bf16 ; also emit dt (cols<64) as bf16
// One wave per 16x16 output tile.
// ---------------------------------------------------------------------------
__global__ __launch_bounds__(32) void gemm2_kernel(
    const bf16* __restrict__ A, const bf16* __restrict__ WT,
    float* __restrict__ xdbl, bf16* __restrict__ dtb) {
  const int lane = threadIdx.x & 31;
  const int m0 = blockIdx.y * 16;
  const int n0 = blockIdx.x * 16;
  v8f acc = {};
  for (int k0 = 0; k0 < ND; k0 += 32) {
    v16bf a = load_A_frag(A, ND, m0, k0, lane);
    v16bf b = load_B_frag(WT, ND, k0, n0, lane);
    acc = wmma_bf16(a, b, acc);
  }
  const int m = m0 + ((lane >> 4) << 3);
  const int n = n0 + (lane & 15);
#pragma unroll
  for (int r = 0; r < 8; ++r) {
    const float v = acc[r];
    xdbl[(size_t)(m + r) * NXP + n] = v;
    if (n0 < DTRANK) dtb[(size_t)(m + r) * DTRANK + n] = (bf16)v;
  }
}

// ---------------------------------------------------------------------------
// GEMM3: delta[16384x1024] = softplus(dt_bf16 @ Wdt_bf16 + bdt)
// ---------------------------------------------------------------------------
__global__ __launch_bounds__(32) void gemm3_kernel(
    const bf16* __restrict__ A, const bf16* __restrict__ WT,
    const float* __restrict__ bias, float* __restrict__ delta) {
  const int lane = threadIdx.x & 31;
  const int m0 = blockIdx.y * 16;
  const int n0 = blockIdx.x * 16;
  v8f acc = {};
  for (int k0 = 0; k0 < DTRANK; k0 += 32) {
    v16bf a = load_A_frag(A, DTRANK, m0, k0, lane);
    v16bf b = load_B_frag(WT, DTRANK, k0, n0, lane);
    acc = wmma_bf16(a, b, acc);
  }
  const int m = m0 + ((lane >> 4) << 3);
  const int n = n0 + (lane & 15);
  const float bv = bias[n];
#pragma unroll
  for (int r = 0; r < 8; ++r) {
    const float x = acc[r] + bv;
    delta[(size_t)(m + r) * ND + n] = fmaxf(x, 0.f) + log1pf(__expf(-fabsf(x)));
  }
}

// ---------------------------------------------------------------------------
// Selective scan: one lane per (b,d,n) state; 16-lane shfl_xor reduce per step
// ---------------------------------------------------------------------------
__global__ __launch_bounds__(256) void scan_kernel(
    const float* __restrict__ delta, const float* __restrict__ x,
    const float* __restrict__ xdbl, const float* __restrict__ A_log,
    const float* __restrict__ Dp, float* __restrict__ out) {
  const int t = blockIdx.x * 256 + threadIdx.x;   // 0 .. B*D*16-1
  const int n = t & (NSTATE - 1);
  const int c = t >> 4;                           // channel: b*D + d
  const int b = c >> 10;
  const int d = c & (ND - 1);
  const float An = -__expf(A_log[d * NSTATE + n]);
  const float Dv = Dp[d];
  const size_t row0 = (size_t)b * NL;
  float h = 0.f;
  for (int l = 0; l < NL; ++l) {
    const size_t rl = row0 + l;
    const float* bc = xdbl + rl * NXP;
    __builtin_prefetch(bc + NXP, 0, 0);           // next timestep's B/C row
    const float dlt = delta[rl * ND + d];
    const float xv  = x[rl * ND + d];
    const float Bn  = bc[DTRANK + n];
    const float Cn  = bc[DTRANK + NSTATE + n];
    const float dA  = __expf(dlt * An);
    h = fmaf(dA, h, (dlt * xv) * Bn);
    float p = h * Cn;
    p += __shfl_xor(p, 1, 32);
    p += __shfl_xor(p, 2, 32);
    p += __shfl_xor(p, 4, 32);
    p += __shfl_xor(p, 8, 32);
    if (n == 0) out[rl * ND + d] = p + xv * Dv;
  }
}

// ---------------------------------------------------------------------------
extern "C" void kernel_launch(void* const* d_in, const int* in_sizes, int n_in,
                              void* d_out, int out_size, void* d_ws, size_t ws_size,
                              hipStream_t stream) {
  const float* f      = (const float*)d_in[0];
  const float* ln_g   = (const float*)d_in[1];
  const float* ln_b   = (const float*)d_in[2];
  const float* Wx     = (const float*)d_in[3];
  const float* bx     = (const float*)d_in[4];
  const float* conv_w = (const float*)d_in[5];
  const float* conv_b = (const float*)d_in[6];
  const float* Wxp    = (const float*)d_in[7];
  const float* Wdt    = (const float*)d_in[8];
  const float* bdt    = (const float*)d_in[9];
  const float* A_log  = (const float*)d_in[10];
  const float* Dp     = (const float*)d_in[11];
  float* out = (float*)d_out;

  // Workspace layout (regions A and P are reused once each, stream-ordered).
  char* ws = (char*)d_ws;
  bf16*  regA  = (bf16*)(ws);                         // 32MB: fn_bf16 -> x_bf16
  bf16*  WxT   = (bf16*)(ws + 33554432);              //  2MB
  bf16*  WxpT  = (bf16*)(ws + 35651584);              // 192KB
  bf16*  WdtT  = (bf16*)(ws + 35848192);              // 128KB
  float* regP  = (float*)(ws + 35979264);             // 64MB: x_pre -> delta
  float* xf32  = (float*)(ws + 103088128);            // 64MB
  float* xdbl  = (float*)(ws + 170196992);            //  6MB
  bf16*  dtb   = (bf16*)(ws + 176488448);             //  2MB

  // 1) Weights: f32 [K][N] -> bf16 K-major [N][K]
  transpose_to_bf16<<<(ND * ND + 255) / 256, 256, 0, stream>>>(Wx, WxT, ND, ND);
  transpose_to_bf16<<<(ND * NXP + 255) / 256, 256, 0, stream>>>(Wxp, WxpT, ND, NXP);
  transpose_to_bf16<<<(DTRANK * ND + 255) / 256, 256, 0, stream>>>(Wdt, WdtT, DTRANK, ND);

  // 2) LayerNorm -> bf16 activations
  layernorm_bf16_kernel<<<NROWS, 256, 0, stream>>>(f, ln_g, ln_b, regA);

  // 3) GEMM1 (WMMA + async-to-LDS double buffering): x_pre = fn @ Wx + bx
  gemm1_kernel<<<dim3(ND / 64, NROWS / 64), 128, 0, stream>>>(regA, WxT, bx, regP);

  // 4) Causal conv + SiLU -> x (f32) and x (bf16, reuses region A)
  conv_silu_kernel<<<16384, 256, 0, stream>>>(regP, conv_w, conv_b, xf32, regA);

  // 5) GEMM2 (WMMA): x_dbl = x @ Wxp ; dt slice to bf16
  gemm2_kernel<<<dim3(NXP / 16, NROWS / 16), 32, 0, stream>>>(regA, WxpT, xdbl, dtb);

  // 6) GEMM3 (WMMA): delta = softplus(dt @ Wdt + bdt)  (reuses region P)
  gemm3_kernel<<<dim3(ND / 16, NROWS / 16), 32, 0, stream>>>(dtb, WdtT, bdt, regP);

  // 7) Selective scan + skip connection
  scan_kernel<<<(NB * ND * NSTATE) / 256, 256, 0, stream>>>(
      regP, xf32, xdbl, A_log, Dp, out);
}
